// MultipleHeadSelfAttention_46978352283891
// MI455X (gfx1250) — matrix operand order repaired and checked by reading, not compiled, not run on hardware
//
#include <hip/hip_runtime.h>
#include <hip/hip_bf16.h>

typedef __bf16 bf16;
typedef __attribute__((ext_vector_type(16))) __bf16 v16bf;
typedef __attribute__((ext_vector_type(8)))  float  v8f;

#define BATCH    64
#define SEQ      256
#define NEMB     384
#define NHEAD    6
#define HDIM     64
#define C3       1152

// Contiguous 32-byte fragment load (two adjacent b128s, no packing moves).
__device__ __forceinline__ v16bf load_frag(const bf16* p) {
  return *(const v16bf*)p;
}

// chunk mapping is the identity: A-tiles are stored linearly so the WMMA
// A-fragment is a contiguous read in contiguous k order, matching B.
__device__ __forceinline__ int chunk_swz(int ch) {
  return ch;
}

// CDNA5 async DMA: global -> LDS, 16 bytes per lane, tracked by ASYNCcnt.
__device__ __forceinline__ void async_b128(bf16* lds, const bf16* g) {
  unsigned l = (unsigned)(uintptr_t)lds;   // low 32 bits = LDS byte offset
  asm volatile("global_load_async_to_lds_b128 %0, %1, off"
               :: "v"(l), "v"(g) : "memory");
}
__device__ __forceinline__ void wait_async_le3() {
  asm volatile("s_wait_asynccnt 0x3" ::: "memory");
}
__device__ __forceinline__ void wait_async_0() {
  asm volatile("s_wait_asynccnt 0x0" ::: "memory");
}

// ---------------------------------------------------------------- prep ------
__global__ void cvt_f32_bf16(const float* __restrict__ x, bf16* __restrict__ y,
                             int n) {
  int i = blockIdx.x * blockDim.x + threadIdx.x;
  if (i < n) y[i] = (bf16)x[i];
}

// w: [K][N] row-major fp32  ->  wt: [N][K] row-major bf16
__global__ void transpose_cvt(const float* __restrict__ w, bf16* __restrict__ wt,
                              int K, int N) {
  int i = blockIdx.x * blockDim.x + threadIdx.x;
  if (i < K * N) {
    int kk = i / N, n = i % N;
    wt[(size_t)n * K + kk] = (bf16)w[(size_t)kk * N + n];
  }
}

// ---------------------------------------------------------------- GEMM ------
// D[M][N] = A[M][K] * Bt[N][K]^T ; block tile 64x128, 8 waves, wave tile 32x32.
// Double-buffered async-to-LDS staging; A slab stored linearly.
template <bool OUT_F32>
__global__ __launch_bounds__(256) void gemm_bf16_wmma(
    const bf16* __restrict__ A, const bf16* __restrict__ Bt,
    void* __restrict__ Dp, int M, int N, int K) {
  __shared__ __align__(32) bf16 sA[2][64 * 32];
  __shared__ __align__(32) bf16 sB[2][128 * 32];
  const int tid  = threadIdx.x;
  const int wave = tid >> 5, lane = tid & 31;
  const int ln = lane & 15, hi = lane >> 4;
  const int wm = wave & 1, wn = wave >> 1;
  const int bm = blockIdx.y * 64;
  const int bn = blockIdx.x * 128;

  v8f acc[2][2] = {};

  const int ar  = tid >> 2;                  // staging row 0..63
  const int ch  = tid & 3;                   // 8-elem chunk 0..3
  const int ac  = ch * 8;                    // source col
  const int acs = chunk_swz(ch) * 8;         // LDS col (A only; same as ac)
  const int nK  = K >> 5;

  auto issue = [&](int buf, int k0) {
    async_b128(&sA[buf][ar * 32 + acs],        A  + (size_t)(bm + ar) * K + k0 + ac);
    async_b128(&sB[buf][ar * 32 + ac],         Bt + (size_t)(bn + ar) * K + k0 + ac);
    async_b128(&sB[buf][(64 + ar) * 32 + ac],  Bt + (size_t)(bn + 64 + ar) * K + k0 + ac);
  };

  issue(0, 0);
  for (int kk = 0; kk < nK; ++kk) {
    const int cur = kk & 1;
    __syncthreads();                         // prior reads of next buffer done
    if (kk + 1 < nK) {
      issue(cur ^ 1, (kk + 1) << 5);
      wait_async_le3();                      // retire current buffer's 3 DMAs
    } else {
      wait_async_0();
    }
    __syncthreads();                         // all waves' DMA writes visible

    v16bf af[2], bfv[2];
#pragma unroll
    for (int mi = 0; mi < 2; ++mi)
      af[mi] = load_frag(sA[cur] + (wm * 32 + mi * 16 + ln) * 32 + hi * 16);
#pragma unroll
    for (int ni = 0; ni < 2; ++ni)
      bfv[ni] = load_frag(sB[cur] + (wn * 32 + ni * 16 + ln) * 32 + hi * 16);
#pragma unroll
    for (int mi = 0; mi < 2; ++mi)
#pragma unroll
      for (int ni = 0; ni < 2; ++ni)
        acc[mi][ni] = __builtin_amdgcn_wmma_f32_16x16x32_bf16(
            false, af[mi], false, bfv[ni], (short)0, acc[mi][ni], false, false);
  }

#pragma unroll
  for (int mi = 0; mi < 2; ++mi)
#pragma unroll
    for (int ni = 0; ni < 2; ++ni)
#pragma unroll
      for (int r = 0; r < 8; ++r) {
        const int row = bm + wm * 32 + mi * 16 + hi * 8 + r;
        const int col = bn + wn * 32 + ni * 16 + ln;
        if constexpr (OUT_F32)
          ((float*)Dp)[(size_t)row * N + col] = acc[mi][ni][r];
        else
          ((bf16*)Dp)[(size_t)row * N + col] = (bf16)acc[mi][ni][r];
      }
}

// ---------------------------------------------------------------- RoPE ------
// qkv: [B*T][1152] bf16 -> q (roped, linear cols) [B,NH,T,64]
//                         k (roped, linear)               [B,NH,T,64]
//                         vT                              [B,NH,64,T]
__global__ void rope_scatter(const bf16* __restrict__ qkv, bf16* __restrict__ q,
                             bf16* __restrict__ k, bf16* __restrict__ vT) {
  int idx = blockIdx.x * blockDim.x + threadIdx.x;
  if (idx >= BATCH * SEQ * NHEAD * (HDIM / 2)) return;
  const int i = idx & 31;               // pair index 0..31
  int tmp = idx >> 5;
  const int h = tmp % NHEAD; tmp /= NHEAD;
  const int t = tmp & (SEQ - 1);
  const int b = tmp >> 8;
  const int d = 2 * i;

  const bf16* base = qkv + (size_t)(b * SEQ + t) * C3 + h * HDIM + d;
  float q0 = (float)base[0],            q1 = (float)base[1];
  float k0 = (float)base[NEMB],         k1 = (float)base[NEMB + 1];
  float v0 = (float)base[2 * NEMB],     v1 = (float)base[2 * NEMB + 1];

  // 10000^(-d/64) = exp(-d * ln(10000)/64)
  const float inv = __expf((float)d * -0.14391156831212787f);
  const float fr  = (float)t * inv;
  float s, c;
  sincosf(fr, &s, &c);

  const size_t ob = ((size_t)(b * NHEAD + h) * SEQ + t) * HDIM;
  // linear column for q (dq == d) (A-operand of S GEMM)
  const int g = d >> 5, e5 = d & 31;
  const int dq = g * 32 + chunk_swz(e5 >> 3) * 8 + (e5 & 7);
  q[ob + dq]     = (bf16)(q0 * c - q1 * s);
  q[ob + dq + 1] = (bf16)(q1 * c + q0 * s);
  k[ob + d]      = (bf16)(k0 * c - k1 * s);
  k[ob + d + 1]  = (bf16)(k1 * c + k0 * s);

  const size_t vb = ((size_t)(b * NHEAD + h) * HDIM + d) * SEQ + t;
  vT[vb]       = (bf16)v0;
  vT[vb + SEQ] = (bf16)v1;
}

// ----------------------------------------------------------- attention ------
// Flash-style causal attention. 1 block per (b,h); 8 waves; each wave owns
// q-tiles {w, 15-w} (balanced). Online softmax over 32-key chunks: runtime
// SGPR loop over fully-valid chunks + one peeled diagonal chunk with element
// masking. Registers: ~2 S tiles + 4 O tiles (no 16-tile score residency).
__global__ __launch_bounds__(256) void attn_wmma(
    const bf16* __restrict__ qg, const bf16* __restrict__ kg,
    const bf16* __restrict__ vTg, bf16* __restrict__ og) {
  __shared__ __align__(32) bf16 sP[8][16 * 32];   // per-wave P strip, 1 KB
  const int bh = blockIdx.x;
  const int b = bh / NHEAD, h = bh % NHEAD;
  const int tid  = threadIdx.x;
  const int wave = tid >> 5, lane = tid & 31;
  const int ln = lane & 15, hi = lane >> 4;
  const bf16* qh = qg  + (size_t)bh * SEQ * HDIM;
  const bf16* kh = kg  + (size_t)bh * SEQ * HDIM;
  const bf16* vh = vTg + (size_t)bh * HDIM * SEQ;
  bf16* sp = &sP[wave][0];

  for (int qi = 0; qi < 2; ++qi) {
    const int qt  = qi ? (15 - wave) : wave;             // balanced pairing
    const int qtu = __builtin_amdgcn_readfirstlane(qt);

    v16bf aq[2];
#pragma unroll
    for (int kk = 0; kk < 2; ++kk)   // q stored linear -> contiguous read
      aq[kk] = load_frag(qh + (size_t)(qt * 16 + ln) * HDIM + kk * 32 + hi * 16);

    float m[8], l[8];
#pragma unroll
    for (int r = 0; r < 8; ++r) { m[r] = -1e30f; l[r] = 0.f; }
    v8f oc[4] = {};

    // one 32-key chunk: S -> online softmax -> P (LDS) -> O += P*V
    auto chunk = [&](int c, bool domask) {
      v8f s0 = {}, s1 = {};
      const bf16* kb0 = kh + (size_t)(c * 32 + ln) * HDIM;
      const bf16* kb1 = kb0 + (size_t)16 * HDIM;
#pragma unroll
      for (int kk = 0; kk < 2; ++kk) {
        s0 = __builtin_amdgcn_wmma_f32_16x16x32_bf16(
            false, aq[kk], false, load_frag(kb0 + kk * 32 + hi * 16),
            (short)0, s0, false, false);
        s1 = __builtin_amdgcn_wmma_f32_16x16x32_bf16(
            false, aq[kk], false, load_frag(kb1 + kk * 32 + hi * 16),
            (short)0, s1, false, false);
      }
      float corr[8];
#pragma unroll
      for (int r = 0; r < 8; ++r) {
        float a0 = s0[r] * 0.125f;                      // 1/sqrt(64)
        float a1 = s1[r] * 0.125f;
        if (domask) {
          const int qrow = qt * 16 + hi * 8 + r;
          a0 = (c * 32 + ln      <= qrow) ? a0 : -1e30f;
          a1 = (c * 32 + 16 + ln <= qrow) ? a1 : -1e30f;
        }
        // chunk row-max across the 16 lanes of this half-wave
        float mt = fmaxf(a0, a1);
#pragma unroll
        for (int off = 1; off < 16; off <<= 1)
          mt = fmaxf(mt, __shfl_xor(mt, off, 32));
        const float mn = fmaxf(m[r], mt);
        corr[r] = __expf(m[r] - mn);
        m[r] = mn;
        const float p0 = __expf(a0 - mn);
        const float p1 = __expf(a1 - mn);
        float su = p0 + p1;
#pragma unroll
        for (int off = 1; off < 16; off <<= 1)
          su += __shfl_xor(su, off, 32);
        l[r] = l[r] * corr[r] + su;
        // P strip (linear cols so A-frag below is one contiguous read)
        const int rowoff = (hi * 8 + r) * 32 + ln;
        sp[rowoff]      = (bf16)p0;
        sp[rowoff + 16] = (bf16)p1;
      }
#pragma unroll
      for (int ot = 0; ot < 4; ++ot)
#pragma unroll
        for (int r = 0; r < 8; ++r)
          oc[ot][r] *= corr[r];
      v16bf ap = load_frag(sp + (size_t)ln * 32 + hi * 16);
#pragma unroll
      for (int ot = 0; ot < 4; ++ot) {
        v16bf bv = load_frag(vh + (size_t)(ot * 16 + ln) * SEQ + c * 32 + hi * 16);
        oc[ot] = __builtin_amdgcn_wmma_f32_16x16x32_bf16(
            false, ap, false, bv, (short)0, oc[ot], false, false);
      }
    };

    const int cd = qtu >> 1;                 // diagonal chunk index
    for (int c = 0; c < cd; ++c)             // fully-valid chunks (SGPR loop)
      chunk(c, false);
    chunk(cd, true);                         // diagonal chunk, element-masked

    // normalize and store to attn-out [B*T][384]
    float invl[8];
#pragma unroll
    for (int r = 0; r < 8; ++r) invl[r] = 1.0f / l[r];
#pragma unroll
    for (int ot = 0; ot < 4; ++ot)
#pragma unroll
      for (int r = 0; r < 8; ++r) {
        const int row = b * SEQ + qt * 16 + hi * 8 + r;
        const int col = h * HDIM + ot * 16 + ln;
        og[(size_t)row * NEMB + col] = (bf16)(oc[ot][r] * invl[r]);
      }
  }
}

// --------------------------------------------------------------- launch -----
extern "C" void kernel_launch(void* const* d_in, const int* in_sizes, int n_in,
                              void* d_out, int out_size, void* d_ws, size_t ws_size,
                              hipStream_t stream) {
  (void)in_sizes; (void)n_in; (void)out_size; (void)ws_size;
  const float* x      = (const float*)d_in[0];
  const float* w_attn = (const float*)d_in[1];
  const float* w_proj = (const float*)d_in[2];
  float* out = (float*)d_out;

  const size_t NX = (size_t)BATCH * SEQ * NEMB;     // 6,291,456
  char* ws = (char*)d_ws;
  bf16* xb   = (bf16*)ws; ws += NX * 2;                       // x in bf16
  bf16* watT = (bf16*)ws; ws += (size_t)C3 * NEMB * 2;        // w_attn^T
  bf16* wpT  = (bf16*)ws; ws += (size_t)NEMB * NEMB * 2;      // w_proj^T
  bf16* qkvb = (bf16*)ws; ws += (size_t)BATCH * SEQ * C3 * 2; // qkv bf16
  bf16* qb   = (bf16*)ws; ws += NX * 2;                       // roped Q (linear)
  bf16* kb   = (bf16*)ws; ws += NX * 2;                       // roped K
  bf16* vTb  = (bf16*)ws; ws += NX * 2;                       // V transposed
  bf16* ob   = (bf16*)ws; ws += NX * 2;                       // attn out bf16

  cvt_f32_bf16<<<(int)(NX / 256), 256, 0, stream>>>(x, xb, (int)NX);
  transpose_cvt<<<(C3 * NEMB + 255) / 256, 256, 0, stream>>>(w_attn, watT, NEMB, C3);
  transpose_cvt<<<(NEMB * NEMB + 255) / 256, 256, 0, stream>>>(w_proj, wpT, NEMB, NEMB);

  // QKV projection: [16384 x 384] x [384 x 1152]
  gemm_bf16_wmma<false><<<dim3(C3 / 128, (BATCH * SEQ) / 64), 256, 0, stream>>>(
      xb, watT, qkvb, BATCH * SEQ, C3, NEMB);

  rope_scatter<<<(BATCH * SEQ * NHEAD * (HDIM / 2)) / 256, 256, 0, stream>>>(
      qkvb, qb, kb, vTb);

  attn_wmma<<<BATCH * NHEAD, 256, 0, stream>>>(qb, kb, vTb, ob);

  // Output projection: [16384 x 384] x [384 x 384] -> fp32 out
  gemm_bf16_wmma<true><<<dim3(NEMB / 128, (BATCH * SEQ) / 64), 256, 0, stream>>>(
      ob, wpT, out, BATCH * SEQ, NEMB, NEMB);
}
